// EncoderBlock_74397423501935
// MI455X (gfx1250) — matrix operand order repaired and checked by reading, not compiled, not run on hardware
//
#include <hip/hip_runtime.h>

// ---------------------------------------------------------------------------
// Types for gfx1250 WMMA (wave32, 16x16x32 bf16 -> f32)
// ---------------------------------------------------------------------------
typedef __attribute__((ext_vector_type(8)))  __bf16 v8bf;
typedef __attribute__((ext_vector_type(16))) __bf16 v16bf;
typedef __attribute__((ext_vector_type(8)))  float  v8f;
typedef __attribute__((ext_vector_type(4)))  unsigned int v4u;
typedef __attribute__((ext_vector_type(8)))  int v8i;
typedef __attribute__((ext_vector_type(4)))  int v4i;

#define CAT16(lo, hi) __builtin_shufflevector(lo, hi, 0,1,2,3,4,5,6,7,8,9,10,11,12,13,14,15)

static __device__ inline v8f vzero8f() {
  v8f z = {0.f, 0.f, 0.f, 0.f, 0.f, 0.f, 0.f, 0.f};
  return z;
}

static __device__ inline v8f wmma_bf16(v16bf a, v16bf b, v8f c) {
  // D = A(16x32 bf16) x B(32x16 bf16) + C(16x16 f32)
  return __builtin_amdgcn_wmma_f32_16x16x32_bf16(false, a, false, b, (short)0, c, false, false);
}

// A-matrix 16x32 (row-major, leading dim lda):
//   lane l<16 : row l, K=0..15
//   lane l>=16: row l-16, K=16..31
static __device__ inline v16bf load_a16x32(const __bf16* tile, int lda, int lane) {
  const __bf16* p = tile + (size_t)(lane & 15) * lda + ((lane >> 4) << 4);
  v8bf lo = *(const v8bf*)(p);
  v8bf hi = *(const v8bf*)(p + 8);
  return CAT16(lo, hi);
}

// B-matrix 32x16 given as Bt[N][K] (N rows, K contiguous, leading dim ldb):
//   lane l<16 : column l, K=0..15 ; lane l>=16: column l-16, K=16..31
static __device__ inline v16bf load_b32x16(const __bf16* tile, int ldb, int lane) {
  const __bf16* p = tile + (size_t)(lane & 15) * ldb + ((lane >> 4) << 4);
  v8bf lo = *(const v8bf*)(p);
  v8bf hi = *(const v8bf*)(p + 8);
  return CAT16(lo, hi);
}

// C/D 16x16 f32: VGPR e, lane l<16 -> (M=e, N=l); lane l>=16 -> (M=e+8, N=l-16)

// ---------------------------------------------------------------------------
// Tensor Data Mover: stage a 128-row x 32-col bf16 panel (row-major, leading
// dim `ld` elements) into LDS at byte offset lds_off. TDM pads each 64-byte
// row (16 DWORDs, pad_interval=3) with 4 DWORDs (pad_amount=3) so the staged
// row stride is 80 B = 40 halves -> 16 rows map to distinct LDS bank groups
// (20*r mod 64 distinct for r=0..15): conflict-free WMMA operand loads.
// D# packing per CDNA5 ISA.
// ---------------------------------------------------------------------------
static __device__ inline void tdm_load_panel(const __bf16* g, unsigned lds_off, int ld) {
  unsigned long long ga = (unsigned long long)g;
  v4u g0;
  g0[0] = 1u;                                            // count=1, user descriptor
  g0[1] = lds_off;                                       // lds_addr
  g0[2] = (unsigned)(ga & 0xFFFFFFFFu);                  // global_addr[31:0]
  g0[3] = (unsigned)((ga >> 32) & 0x01FFFFFFu) | (2u << 30); // addr[56:32] | type=2
  v8i g1;
  g1[0] = (1 << 16) | (1 << 20) | (3 << 22) | (3 << 25); // data_size=2B, pad_en, int=3, amt=3
  g1[1] = (32 << 16);                                    // tensor_dim0 = 32 (lo16 in [31:16])
  g1[2] = (128 << 16);                                   // dim0 hi=0 | tensor_dim1 = 128
  g1[3] = (32 << 16);                                    // dim1 hi=0 | tile_dim0 = 32
  g1[4] = 128;                                           // tile_dim1 = 128, tile_dim2 = 0
  g1[5] = ld;                                            // tensor_dim0_stride[31:0]
  g1[6] = 0;                                             // stride hi | dim1_stride lo
  g1[7] = 0;
  v4i z4 = {0, 0, 0, 0};                                 // 2D tensor: groups 2/3 unused
  v8i z8 = {0, 0, 0, 0, 0, 0, 0, 0};
  __builtin_amdgcn_tensor_load_to_lds(g0, g1, z4, z4, z8, 0);
}

static __device__ inline unsigned lds_off_of(const void* p) {
  return (unsigned)(unsigned long long)p;                // flat LDS addr[31:0] = LDS offset
}

// ---------------------------------------------------------------------------
// Weight prep: fp32 -> bf16, transposed / zero-padded layouts
// ---------------------------------------------------------------------------
__global__ void prep_qkvw(const float* __restrict__ w, __bf16* __restrict__ wT) {
  // w: [12][512][42]  ->  wT: [12][64][512]  (e-major, rows 42..63 zero)
  int i = blockIdx.x * 256 + threadIdx.x;
  if (i >= 12 * 64 * 512) return;
  int d = i & 511, e = (i >> 9) & 63, h = i >> 15;
  float v = (e < 42) ? w[((size_t)h * 512 + d) * 42 + e] : 0.f;
  wT[i] = (__bf16)v;
}

__global__ void prep_wp(const float* __restrict__ w, __bf16* __restrict__ wT) {
  // w_proj: [504][512] -> wpT: [512][768], wpT[d][h*64+e] = w[h*42+e][d], pad 0
  int i = blockIdx.x * 256 + threadIdx.x;
  if (i >= 512 * 768) return;
  int hk = i % 768, d = i / 768;
  int h = hk >> 6, e = hk & 63;
  float v = (e < 42) ? w[(size_t)(h * 42 + e) * 512 + d] : 0.f;
  wT[i] = (__bf16)v;
}

__global__ void prep_w1(const float* __restrict__ w, __bf16* __restrict__ wT) {
  // w1: [512][2048] -> w1T: [2048][512]
  int i = blockIdx.x * 256 + threadIdx.x;
  if (i >= 2048 * 512) return;
  int d = i & 511, j = i >> 9;
  wT[i] = (__bf16)w[(size_t)d * 2048 + j];
}

__global__ void prep_w2(const float* __restrict__ w, __bf16* __restrict__ wT) {
  // w2: [2048][512] -> w2T: [512][2048]
  int i = blockIdx.x * 256 + threadIdx.x;
  if (i >= 512 * 2048) return;
  int j = i & 2047, d = i >> 11;
  wT[i] = (__bf16)w[(size_t)j * 512 + d];
}

// ---------------------------------------------------------------------------
// LayerNorm (one wave per 512-wide row)  out = LN(x) in bf16
// ---------------------------------------------------------------------------
__global__ __launch_bounds__(256) void ln_rows(const float* __restrict__ x,
                                               const float* __restrict__ gamma,
                                               const float* __restrict__ beta,
                                               __bf16* __restrict__ outb, int nrows) {
  int row = blockIdx.x * 8 + (threadIdx.x >> 5);
  if (row >= nrows) return;
  int lane = threadIdx.x & 31;
  const float* xr = x + (size_t)row * 512;
  float s = 0.f, s2 = 0.f;
  for (int i = lane; i < 512; i += 32) { float v = xr[i]; s += v; s2 += v * v; }
  for (int o = 16; o > 0; o >>= 1) { s += __shfl_xor(s, o); s2 += __shfl_xor(s2, o); }
  float mu = s * (1.f / 512.f);
  float var = s2 * (1.f / 512.f) - mu * mu;
  float rs = rsqrtf(var + 1e-5f);
  for (int i = lane; i < 512; i += 32)
    outb[(size_t)row * 512 + i] = (__bf16)((xr[i] - mu) * rs * gamma[i] + beta[i]);
}

// first_added = attn0[s] + x[b,s]; fa (f32) kept for residual, n2 = LN(fa) in bf16
__global__ __launch_bounds__(256) void add_ln(const float* __restrict__ attn0,
                                              const float* __restrict__ x,
                                              const float* __restrict__ gamma,
                                              const float* __restrict__ beta,
                                              float* __restrict__ fa,
                                              __bf16* __restrict__ n2) {
  int row = blockIdx.x * 8 + (threadIdx.x >> 5);
  if (row >= 8192) return;
  int lane = threadIdx.x & 31;
  int sidx = row & 1023;
  const float* ar = attn0 + (size_t)sidx * 512;
  const float* xr = x + (size_t)row * 512;
  float* fr = fa + (size_t)row * 512;
  float s = 0.f, s2 = 0.f;
  for (int i = lane; i < 512; i += 32) {
    float v = ar[i] + xr[i];
    fr[i] = v; s += v; s2 += v * v;
  }
  for (int o = 16; o > 0; o >>= 1) { s += __shfl_xor(s, o); s2 += __shfl_xor(s2, o); }
  float mu = s * (1.f / 512.f);
  float var = s2 * (1.f / 512.f) - mu * mu;
  float rs = rsqrtf(var + 1e-5f);
  for (int i = lane; i < 512; i += 32)
    n2[(size_t)row * 512 + i] = (__bf16)((fr[i] - mu) * rs * gamma[i] + beta[i]);
}

// ---------------------------------------------------------------------------
// QKV projection (batch 0 only): per head, [1024x512] @ [512x64]
// q,k stored [h][s][64]; v stored transposed [h][64][s] for the PV GEMM
// ---------------------------------------------------------------------------
__global__ __launch_bounds__(256) void qkv_proj(const __bf16* __restrict__ n1,
                                                const __bf16* __restrict__ wqT,
                                                const __bf16* __restrict__ wkT,
                                                const __bf16* __restrict__ wvT,
                                                __bf16* __restrict__ q,
                                                __bf16* __restrict__ k,
                                                __bf16* __restrict__ vT) {
  int lane = threadIdx.x & 31;
  int flat = blockIdx.x * 8 + (threadIdx.x >> 5);
  if (flat >= 3 * 12 * 32) return;
  int which = flat / (12 * 32);
  int rem = flat % (12 * 32);
  int h = rem / 32, tm = rem % 32;
  const __bf16* W = (which == 0 ? wqT : which == 1 ? wkT : wvT) + (size_t)h * 64 * 512;
  v8f acc[2][4];
#pragma unroll
  for (int i = 0; i < 2; ++i)
#pragma unroll
    for (int j = 0; j < 4; ++j) acc[i][j] = vzero8f();

  const __bf16* A0 = n1 + (size_t)(tm * 32) * 512;
  const __bf16* A1 = A0 + (size_t)16 * 512;
  for (int k0 = 0; k0 < 512; k0 += 32) {
    v16bf a0 = load_a16x32(A0 + k0, 512, lane);
    v16bf a1 = load_a16x32(A1 + k0, 512, lane);
#pragma unroll
    for (int j = 0; j < 4; ++j) {
      v16bf b = load_b32x16(W + (size_t)(j * 16) * 512 + k0, 512, lane);
      acc[0][j] = wmma_bf16(a0, b, acc[0][j]);
      acc[1][j] = wmma_bf16(a1, b, acc[1][j]);
    }
  }
  int mrow = (lane >> 4) << 3, ncol = lane & 15;
  __bf16* dst = (which == 0 ? q : which == 1 ? k : vT) + (size_t)h * 64 * 1024;
#pragma unroll
  for (int i = 0; i < 2; ++i)
#pragma unroll
    for (int j = 0; j < 4; ++j)
#pragma unroll
      for (int e = 0; e < 8; ++e) {
        int s = tm * 32 + i * 16 + mrow + e;
        int ec = j * 16 + ncol;
        __bf16 val = (__bf16)acc[i][j][e];
        if (which == 2) dst[(size_t)ec * 1024 + s] = val;   // vT[e][s]
        else            dst[(size_t)s * 64 + ec] = val;     // q/k[s][e]
      }
}

// ---------------------------------------------------------------------------
// Attention for batch 0. One wave per (head, 16-query tile).
// LDS: 16x1024 f32 scores (64KB) + 16x1024 bf16 probs (32KB) = 96KB dynamic.
// ---------------------------------------------------------------------------
__global__ __launch_bounds__(32) void attn_kernel(const __bf16* __restrict__ q,
                                                  const __bf16* __restrict__ k,
                                                  const __bf16* __restrict__ vT,
                                                  __bf16* __restrict__ concat) {
  extern __shared__ char smem[];
  float*  sc = (float*)smem;                    // [16][1024]
  __bf16* pr = (__bf16*)(smem + 16 * 1024 * 4); // [16][1024]
  int lane = threadIdx.x;
  int qt = blockIdx.x;          // 0..63 query tiles
  int h  = blockIdx.y;          // 0..11 heads
  const __bf16* qh = q  + (size_t)h * 1024 * 64;
  const __bf16* kh = k  + (size_t)h * 1024 * 64;
  const __bf16* vh = vT + (size_t)h * 64 * 1024;

  // Q row-tile A operands (reused across all 64 key tiles)
  v16bf a0 = load_a16x32(qh + (size_t)(qt * 16) * 64 + 0, 64, lane);
  v16bf a1 = load_a16x32(qh + (size_t)(qt * 16) * 64 + 32, 64, lane);

  int mrow = (lane >> 4) << 3, ncol = lane & 15;
  const float scale = 0.15430334996209191f;     // 1/sqrt(42)

  // Phase A: scores = (Q K^T) * scale  -> LDS (f32)
  for (int kt = 0; kt < 64; ++kt) {
    v16bf b0 = load_b32x16(kh + (size_t)(kt * 16) * 64 + 0, 64, lane);
    v16bf b1 = load_b32x16(kh + (size_t)(kt * 16) * 64 + 32, 64, lane);
    v8f c = vzero8f();
    c = wmma_bf16(a0, b0, c);
    c = wmma_bf16(a1, b1, c);
#pragma unroll
    for (int e = 0; e < 8; ++e)
      sc[(mrow + e) * 1024 + kt * 16 + ncol] = c[e] * scale;
  }
  __syncthreads();

  // Phase B: softmax per row (2 lanes per row, xor-16 pair reduction)
  int r = lane & 15, par = lane >> 4;
  float m = -3.4e38f;
  for (int c = par; c < 1024; c += 2) m = fmaxf(m, sc[r * 1024 + c]);
  m = fmaxf(m, __shfl_xor(m, 16));
  float sum = 0.f;
  for (int c = par; c < 1024; c += 2) {
    float e = expf(sc[r * 1024 + c] - m);
    sc[r * 1024 + c] = e;
    sum += e;
  }
  sum += __shfl_xor(sum, 16);
  float inv = 1.0f / sum;
  for (int c = par; c < 1024; c += 2)
    pr[r * 1024 + c] = (__bf16)(sc[r * 1024 + c] * inv);
  __syncthreads();

  // Phase C: out = probs[16x1024] @ V[1024x64]; A tiles read from LDS
  v8f acc[4];
#pragma unroll
  for (int j = 0; j < 4; ++j) acc[j] = vzero8f();
  for (int kc = 0; kc < 32; ++kc) {
    v16bf a = load_a16x32(pr + kc * 32, 1024, lane);
#pragma unroll
    for (int j = 0; j < 4; ++j) {
      v16bf b = load_b32x16(vh + (size_t)(j * 16) * 1024 + kc * 32, 1024, lane);
      acc[j] = wmma_bf16(a, b, acc[j]);
    }
  }
  // concat layout [1024][768], head h occupies cols h*64..h*64+63 (pad cols are 0)
#pragma unroll
  for (int j = 0; j < 4; ++j)
#pragma unroll
    for (int e = 0; e < 8; ++e)
      concat[(size_t)(qt * 16 + mrow + e) * 768 + h * 64 + j * 16 + ncol] = (__bf16)acc[j][e];
}

// ---------------------------------------------------------------------------
// TDM-staged WMMA GEMM. Block = 256 threads (8 waves), block tile 128M x 128N.
// Wave 0 double-buffers 128x32 panels of A and Bt into LDS via the Tensor
// Data Mover (TENSORcnt-tracked); all 8 waves compute 32Mx64N sub-tiles with
// 8 WMMAs per 32-wide K step, reading conflict-free 80B-stride LDS rows.
// A: bf16 [MxK] row-major (lda); Bt: bf16 [NxK] row-major (ldb).
// MODE 0: outf = acc + bias
// MODE 1: outh = bf16(gelu_exact(acc + bias))
// MODE 2: outf = acc + bias + resid
// ---------------------------------------------------------------------------
template <int MODE>
__global__ __launch_bounds__(256) void gemm_tdm(const __bf16* __restrict__ A,
                                                const __bf16* __restrict__ Bt,
                                                const float* __restrict__ bias,
                                                const float* __restrict__ resid,
                                                float* __restrict__ outf,
                                                __bf16* __restrict__ outh,
                                                int M, int N, int K,
                                                int lda, int ldb, int ldc) {
  __shared__ __attribute__((aligned(16))) __bf16 Abuf[2][128 * 40]; // 80B row stride
  __shared__ __attribute__((aligned(16))) __bf16 Bbuf[2][128 * 40];
  int lane = threadIdx.x & 31;
  int wave = threadIdx.x >> 5;
  int nbn = N >> 7;
  int bm = blockIdx.x / nbn, bn = blockIdx.x % nbn;
  int wm = wave >> 1, wn = wave & 1;   // 4 x 2 wave grid over the 128x128 tile
  int nk = K >> 5;

  const __bf16* Ag = A + (size_t)(bm * 128) * lda;
  const __bf16* Bg = Bt + (size_t)(bn * 128) * ldb;

  v8f acc[2][4];
#pragma unroll
  for (int i = 0; i < 2; ++i)
#pragma unroll
    for (int j = 0; j < 4; ++j) acc[i][j] = vzero8f();

  if (wave == 0) {
    tdm_load_panel(Ag, lds_off_of(&Abuf[0][0]), lda);
    tdm_load_panel(Bg, lds_off_of(&Bbuf[0][0]), ldb);
  }
  for (int ks = 0; ks < nk; ++ks) {
    int cur = ks & 1;
    if (wave == 0) {
      if (ks + 1 < nk) {
        tdm_load_panel(Ag + (ks + 1) * 32, lds_off_of(&Abuf[cur ^ 1][0]), lda);
        tdm_load_panel(Bg + (ks + 1) * 32, lds_off_of(&Bbuf[cur ^ 1][0]), ldb);
        __builtin_amdgcn_s_wait_tensorcnt(2);  // in-order: current panels done
      } else {
        __builtin_amdgcn_s_wait_tensorcnt(0);
      }
    }
    __syncthreads();                            // panels [cur] visible to all waves
    const __bf16* As = &Abuf[cur][0] + (size_t)(wm * 32) * 40;
    const __bf16* Bs = &Bbuf[cur][0] + (size_t)(wn * 64) * 40;
    v16bf a0 = load_a16x32(As, 40, lane);
    v16bf a1 = load_a16x32(As + 16 * 40, 40, lane);
#pragma unroll
    for (int j = 0; j < 4; ++j) {
      v16bf b = load_b32x16(Bs + (size_t)(j * 16) * 40, 40, lane);
      acc[0][j] = wmma_bf16(a0, b, acc[0][j]);
      acc[1][j] = wmma_bf16(a1, b, acc[1][j]);
    }
    __syncthreads();                            // done reading before TDM overwrites
  }

  int mrow = (lane >> 4) << 3, ncol = lane & 15;
#pragma unroll
  for (int i = 0; i < 2; ++i)
#pragma unroll
    for (int j = 0; j < 4; ++j)
#pragma unroll
      for (int e = 0; e < 8; ++e) {
        int row = bm * 128 + wm * 32 + i * 16 + mrow + e;
        int col = bn * 128 + wn * 64 + j * 16 + ncol;
        float v = acc[i][j][e] + bias[col];
        if (MODE == 0) {
          outf[(size_t)row * ldc + col] = v;
        } else if (MODE == 1) {
          float g = 0.5f * v * (1.0f + erff(v * 0.70710678118654752f));
          outh[(size_t)row * ldc + col] = (__bf16)g;
        } else {
          outf[(size_t)row * ldc + col] = v + resid[(size_t)row * ldc + col];
        }
      }
}

// ---------------------------------------------------------------------------
// Host launcher
// ---------------------------------------------------------------------------
extern "C" void kernel_launch(void* const* d_in, const int* in_sizes, int n_in,
                              void* d_out, int out_size, void* d_ws, size_t ws_size,
                              hipStream_t stream) {
  (void)in_sizes; (void)n_in; (void)out_size; (void)ws_size;
  const float* x      = (const float*)d_in[0];   // [8,1024,512]
  const float* wk     = (const float*)d_in[1];   // [12,512,42]
  const float* wq     = (const float*)d_in[2];   // [12,512,42]
  const float* wv     = (const float*)d_in[3];   // [12,512,42]
  const float* w_proj = (const float*)d_in[4];   // [504,512]
  const float* b_proj = (const float*)d_in[5];   // [512]
  const float* gamma  = (const float*)d_in[6];   // [512]
  const float* beta   = (const float*)d_in[7];   // [512]
  const float* w1     = (const float*)d_in[8];   // [512,2048]
  const float* b1     = (const float*)d_in[9];   // [2048]
  const float* w2     = (const float*)d_in[10];  // [2048,512]
  const float* b2     = (const float*)d_in[11];  // [512]
  float* out = (float*)d_out;                    // [8,1024,512]

  char* ws = (char*)d_ws;
  size_t off = 0;
  auto alloc = [&](size_t bytes) {
    char* p = ws + off;
    off = (off + bytes + 255) & ~(size_t)255;
    return p;
  };
  __bf16* n1_0   = (__bf16*)alloc((size_t)1024 * 512 * 2);
  __bf16* wqT    = (__bf16*)alloc((size_t)12 * 64 * 512 * 2);
  __bf16* wkT    = (__bf16*)alloc((size_t)12 * 64 * 512 * 2);
  __bf16* wvT    = (__bf16*)alloc((size_t)12 * 64 * 512 * 2);
  __bf16* qb     = (__bf16*)alloc((size_t)12 * 1024 * 64 * 2);
  __bf16* kb     = (__bf16*)alloc((size_t)12 * 1024 * 64 * 2);
  __bf16* vTb    = (__bf16*)alloc((size_t)12 * 64 * 1024 * 2);
  __bf16* wpT    = (__bf16*)alloc((size_t)512 * 768 * 2);
  __bf16* concat = (__bf16*)alloc((size_t)1024 * 768 * 2);
  float*  attn0  = (float*) alloc((size_t)1024 * 512 * 4);
  float*  fa     = (float*) alloc((size_t)8 * 1024 * 512 * 4);
  __bf16* n2     = (__bf16*)alloc((size_t)8 * 1024 * 512 * 2);
  __bf16* w1T    = (__bf16*)alloc((size_t)2048 * 512 * 2);
  __bf16* w2T    = (__bf16*)alloc((size_t)512 * 2048 * 2);
  __bf16* hbuf   = (__bf16*)alloc((size_t)8192 * 2048 * 2);

  // Weight conversion (bf16, transposed/padded layouts)
  prep_qkvw<<<1536, 256, 0, stream>>>(wq, wqT);
  prep_qkvw<<<1536, 256, 0, stream>>>(wk, wkT);
  prep_qkvw<<<1536, 256, 0, stream>>>(wv, wvT);
  prep_wp  <<<1536, 256, 0, stream>>>(w_proj, wpT);
  prep_w1  <<<4096, 256, 0, stream>>>(w1, w1T);
  prep_w2  <<<4096, 256, 0, stream>>>(w2, w2T);

  // LN1 — only batch 0 feeds the attention path (reference uses attn[0] only)
  ln_rows<<<128, 256, 0, stream>>>(x, gamma, beta, n1_0, 1024);

  // Q/K/V (batch 0): 3*12 GEMMs of [1024x512]@[512x64]
  qkv_proj<<<144, 256, 0, stream>>>(n1_0, wqT, wkT, wvT, qb, kb, vTb);

  // Attention (batch 0): 12 heads x 64 query tiles, 96KB dynamic LDS per wave-block
  attn_kernel<<<dim3(64, 12), 32, 96 * 1024, stream>>>(qb, kb, vTb, concat);

  // Output projection: [1024x768]@[768x512] + b_proj -> attn0 (f32), TDM-staged
  gemm_tdm<0><<<8 * 4, 256, 0, stream>>>(concat, wpT, b_proj, nullptr, attn0, nullptr,
                                         1024, 512, 768, 768, 768, 512);

  // first_added = attn0 (broadcast) + x; LN2 -> n2 (bf16)
  add_ln<<<1024, 256, 0, stream>>>(attn0, x, gamma, beta, fa, n2);

  // FFN1: [8192x512]@[512x2048] + b1, exact GELU -> hbuf (bf16), TDM-staged
  gemm_tdm<1><<<64 * 16, 256, 0, stream>>>(n2, w1T, b1, nullptr, nullptr, hbuf,
                                           8192, 2048, 512, 512, 512, 2048);

  // FFN2: [8192x2048]@[2048x512] + b2 + first_added -> out (f32), TDM-staged
  gemm_tdm<2><<<64 * 4, 256, 0, stream>>>(hbuf, w2T, b2, fa, out, nullptr,
                                          8192, 512, 2048, 2048, 2048, 512);
}
